// CausalSelfAttention_72739566125774
// MI455X (gfx1250) — matrix operand order repaired and checked by reading, not compiled, not run on hardware
//
#include <hip/hip_runtime.h>

// ---------------------------------------------------------------------------
// CDNA5 (gfx1250, wave32) causal self-attention, all matmuls via
// v_wmma_f32_16x16x32_bf16.  A one-shot prep kernel converts fp32 inputs to
// bf16 in workspace so the GEMM inner loops are pure bf16 b128 loads + WMMA.
// GEMM waves compute 32x64 tiles (8 WMMA per k-step, shared B fragments).
// Wave-uniform indices are forced into SGPRs via readfirstlane so epilogue
// address math (div by 768/12) is scalar, not per-lane VALU.
// ---------------------------------------------------------------------------

typedef __attribute__((ext_vector_type(16))) __bf16 bf16x16;
typedef __attribute__((ext_vector_type(8)))  __bf16 bf16x8;
typedef __attribute__((ext_vector_type(8)))  float  f32x8;

#define BATCH 8
#define NHEAD 12
#define TSEQ  1024
#define HD    64
#define CEMB  768

__device__ __forceinline__ __bf16 f2bf(float f) { return (__bf16)f; }

__device__ __forceinline__ f32x8 wmma_bf16(bf16x16 a, bf16x16 b, f32x8 c) {
  return __builtin_amdgcn_wmma_f32_16x16x32_bf16(
      /*neg_a=*/false, a, /*neg_b=*/false, b,
      /*c_mod=*/(short)0, c, /*reuse_a=*/false, /*reuse_b=*/false);
}

// A fragment (16x32, M x K).  lane&15 = row; lanes<16 hold K = k0+0..7 and
// k0+16..23, lanes>=16 hold K = k0+8..15 and k0+24..31 (ISA 7.12.2).
__device__ __forceinline__ bf16x16 a_frag_bf16(const __bf16* rowp, int k0, int lane) {
  int kb = k0 + ((lane & 16) ? 8 : 0);
  bf16x16 a;
#pragma unroll
  for (int i = 0; i < 8; ++i) a[i] = rowp[kb + i];
#pragma unroll
  for (int i = 0; i < 8; ++i) a[8 + i] = rowp[kb + 16 + i];
  return a;
}

// B fragment (32x16, K x N).  colp points at column (lane&15); lanes<16 hold
// K = k0+0..7 and k0+16..23, lanes>=16 hold K = k0+8..15 and k0+24..31.
__device__ __forceinline__ bf16x16 b_frag_bf16(const __bf16* colp, int k0, int lane) {
  int kb = k0 + ((lane & 16) ? 8 : 0);
  bf16x16 b;
#pragma unroll
  for (int i = 0; i < 16; ++i) b[i] = colp[kb + i + (i & 8)];
  return b;
}

// cross-lane row reductions inside each 16-lane half (wave32)
__device__ __forceinline__ float rowmax16(float v) {
#pragma unroll
  for (int m = 1; m < 16; m <<= 1) v = fmaxf(v, __shfl_xor(v, m, 32));
  return v;
}
__device__ __forceinline__ float rowsum16(float v) {
#pragma unroll
  for (int m = 1; m < 16; m <<= 1) v += __shfl_xor(v, m, 32);
  return v;
}

// ---------------------------------------------------------------------------
// Kernel 0: one-shot fp32 -> bf16 conversion (8 elements/thread, 16B stores).
// ---------------------------------------------------------------------------
__global__ __launch_bounds__(256) void cvt_kernel(const float* __restrict__ src,
                                                  __bf16* __restrict__ dst, int n) {
  int i = (blockIdx.x * 256 + threadIdx.x) * 8;
  if (i < n) {
    bf16x8 v;
#pragma unroll
    for (int j = 0; j < 8; ++j) v[j] = f2bf(src[i + j]);
    *(bf16x8*)(dst + i) = v;
  }
}

// ---------------------------------------------------------------------------
// Kernel 1: qkv = x @ w_attn^T + b_attn, scattered to Q[bh,t,d], K[bh,t,d],
// V^T[bh,d,t] in bf16.  M=8192, N=2304, K=768.  Wave computes 32x64.
// grid = (256, 9), block = 128 (4 waves, wave w -> N offset 64*w).
// ---------------------------------------------------------------------------
__global__ __launch_bounds__(128) void qkv_kernel(
    const __bf16* __restrict__ X, const __bf16* __restrict__ Wa,
    const float* __restrict__ Ba,
    __bf16* __restrict__ Q, __bf16* __restrict__ K, __bf16* __restrict__ VT) {
  int lane = threadIdx.x & 31;
  int wave = __builtin_amdgcn_readfirstlane(threadIdx.x >> 5);  // SGPR
  int m0 = blockIdx.x * 32;
  int n0 = blockIdx.y * 256 + wave * 64;                       // SGPR

  const __bf16* xrow0 = X + (size_t)(m0 + (lane & 15)) * CEMB;
  const __bf16* xrow1 = xrow0 + (size_t)16 * CEMB;

  f32x8 acc[2][4];
#pragma unroll
  for (int i = 0; i < 2; ++i)
#pragma unroll
    for (int j = 0; j < 4; ++j)
#pragma unroll
      for (int r = 0; r < 8; ++r) acc[i][j][r] = 0.0f;

  for (int k0 = 0; k0 < CEMB; k0 += 32) {
    bf16x16 a0 = a_frag_bf16(xrow0, k0, lane);
    bf16x16 a1 = a_frag_bf16(xrow1, k0, lane);
#pragma unroll
    for (int j = 0; j < 4; ++j) {
      const __bf16* wcol = Wa + (size_t)(n0 + j * 16 + (lane & 15)) * CEMB;
      bf16x16 b = b_frag_bf16(wcol, k0, lane);
      acc[0][j] = wmma_bf16(a0, b, acc[0][j]);
      acc[1][j] = wmma_bf16(a1, b, acc[1][j]);
    }
  }

  // epilogue: section/head/batch splits are wave-uniform scalars (tiles are
  // aligned and never cross a 768- or 1024-boundary).
  int bq    = m0 >> 10;         // batch index (scalar)
  int tbase = m0 & 1023;        // t of row 0 (scalar)
  int col   = lane & 15;
  int roff  = (lane & 16) ? 8 : 0;
#pragma unroll
  for (int j = 0; j < 4; ++j) {
    int otile = n0 + j * 16;    // scalar
    int sec = otile / CEMB;     // scalar division
    int cc = otile - sec * CEMB;
    int h = cc >> 6;            // scalar
    int dbase = cc & 63;        // scalar (multiple of 16)
    int bh = bq * NHEAD + h;    // scalar
    int d = dbase + col;
    float bias = Ba[otile + col];
#pragma unroll
    for (int i = 0; i < 2; ++i) {
#pragma unroll
      for (int r = 0; r < 8; ++r) {
        int t = tbase + i * 16 + r + roff;
        __bf16 bv = f2bf(acc[i][j][r] + bias);
        if (sec == 0)      Q[((size_t)bh * TSEQ + t) * HD + d] = bv;
        else if (sec == 1) K[((size_t)bh * TSEQ + t) * HD + d] = bv;
        else               VT[((size_t)bh * HD + d) * TSEQ + t] = bv;
      }
    }
  }
}

// ---------------------------------------------------------------------------
// Kernel 2: flash-style causal attention.  One wave per 16-query tile,
// key blocks of 32 (P tile = 16x32 bf16 A-fragment).  O is 16x64 fp32.
// grid = B*H*(T/16)/4 = 1536 blocks, block = 128 (4 waves).
// ---------------------------------------------------------------------------
__global__ __launch_bounds__(128) void attn_kernel(
    const __bf16* __restrict__ Q, const __bf16* __restrict__ K,
    const __bf16* __restrict__ VT, __bf16* __restrict__ Y) {
  __shared__ __bf16 Plds[4][16 * 32];  // per-wave P transpose staging

  int lane = threadIdx.x & 31;
  int wave = __builtin_amdgcn_readfirstlane(threadIdx.x >> 5);  // SGPR
  int tile = blockIdx.x * 4 + wave;                             // SGPR
  int qt = tile & (TSEQ / 16 - 1);
  int bh = tile / (TSEQ / 16);
  int b = bh / NHEAD, h = bh % NHEAD;                           // scalar div
  int q0 = qt * 16;

  const __bf16* qrow = Q + ((size_t)bh * TSEQ + q0 + (lane & 15)) * HD;
  bf16x16 qa0 = a_frag_bf16(qrow, 0, lane);
  bf16x16 qa1 = a_frag_bf16(qrow, 32, lane);

  float m_r[8], l_r[8];
  f32x8 o[4];
#pragma unroll
  for (int r = 0; r < 8; ++r) { m_r[r] = -1e30f; l_r[r] = 0.0f; }
#pragma unroll
  for (int j = 0; j < 4; ++j)
#pragma unroll
    for (int r = 0; r < 8; ++r) o[j][r] = 0.0f;

  for (int k0 = 0; k0 < q0 + 16; k0 += 32) {
    // ---- S = Q @ K^T for 32 keys (two 16x16 C fragments) ----
    f32x8 s0, s1;
#pragma unroll
    for (int r = 0; r < 8; ++r) { s0[r] = 0.0f; s1[r] = 0.0f; }

    const __bf16* kc0 = K + ((size_t)bh * TSEQ + k0 + (lane & 15)) * HD;
    const __bf16* kc1 = kc0 + (size_t)16 * HD;
    s0 = wmma_bf16(qa0, b_frag_bf16(kc0, 0, lane), s0);
    s0 = wmma_bf16(qa1, b_frag_bf16(kc0, 32, lane), s0);
    s1 = wmma_bf16(qa0, b_frag_bf16(kc1, 0, lane), s1);
    s1 = wmma_bf16(qa1, b_frag_bf16(kc1, 32, lane), s1);

    // ---- scale + causal mask + online softmax ----
    int key0 = k0 + (lane & 15);
    int key1 = key0 + 16;
#pragma unroll
    for (int r = 0; r < 8; ++r) {
      int qi = q0 + r + ((lane & 16) ? 8 : 0);
      float a0 = s0[r] * 0.125f;            // 1/sqrt(64)
      float a1 = s1[r] * 0.125f;
      if (key0 > qi) a0 = -1e30f;
      if (key1 > qi) a1 = -1e30f;

      float mx = rowmax16(fmaxf(a0, a1));
      float mnew = fmaxf(m_r[r], mx);
      float corr = __expf(m_r[r] - mnew);
      float p0 = __expf(a0 - mnew);
      float p1 = __expf(a1 - mnew);
      l_r[r] = l_r[r] * corr + rowsum16(p0 + p1);
      m_r[r] = mnew;
#pragma unroll
      for (int j = 0; j < 4; ++j) o[j][r] *= corr;

      int row = r + ((lane & 16) ? 8 : 0);
      int col = lane & 15;
      Plds[wave][row * 32 + col]      = f2bf(p0);
      Plds[wave][row * 32 + col + 16] = f2bf(p1);
    }

    // wave-private LDS round trip: C-layout -> A-layout transpose
    asm volatile("s_wait_dscnt 0" ::: "memory");
    bf16x16 pa;
    {
      const __bf16* prow = &Plds[wave][(lane & 15) * 32];
      int kb = (lane & 16) ? 8 : 0;
#pragma unroll
      for (int i = 0; i < 8; ++i) pa[i] = prow[kb + i];
#pragma unroll
      for (int i = 0; i < 8; ++i) pa[8 + i] = prow[kb + 16 + i];
    }

    // ---- O += P @ V  (V^T rows are contiguous over keys) ----
#pragma unroll
    for (int j = 0; j < 4; ++j) {
      const __bf16* vrow = VT + ((size_t)bh * HD + j * 16 + (lane & 15)) * TSEQ;
      o[j] = wmma_bf16(pa, b_frag_bf16(vrow, k0, lane), o[j]);
    }
  }

  // ---- normalize and store Y[b, t, h*64 + d] (bf16) ----
#pragma unroll
  for (int j = 0; j < 4; ++j) {
#pragma unroll
    for (int r = 0; r < 8; ++r) {
      int qi = q0 + r + ((lane & 16) ? 8 : 0);
      int d = j * 16 + (lane & 15);
      float val = o[j][r] / l_r[r];
      Y[((size_t)b * TSEQ + qi) * CEMB + h * HD + d] = f2bf(val);
    }
  }
}

// ---------------------------------------------------------------------------
// Kernel 3: out = y @ w_proj^T + b_proj.  M=8192, N=768, K=768.
// Wave computes 32x64.  grid = (256, 3), block = 128.
// ---------------------------------------------------------------------------
__global__ __launch_bounds__(128) void proj_kernel(
    const __bf16* __restrict__ Yb, const __bf16* __restrict__ Wp,
    const float* __restrict__ Bp, float* __restrict__ Out) {
  int lane = threadIdx.x & 31;
  int wave = __builtin_amdgcn_readfirstlane(threadIdx.x >> 5);  // SGPR
  int m0 = blockIdx.x * 32;
  int n0 = blockIdx.y * 256 + wave * 64;                        // SGPR

  const __bf16* yrow0 = Yb + (size_t)(m0 + (lane & 15)) * CEMB;
  const __bf16* yrow1 = yrow0 + (size_t)16 * CEMB;

  f32x8 acc[2][4];
#pragma unroll
  for (int i = 0; i < 2; ++i)
#pragma unroll
    for (int j = 0; j < 4; ++j)
#pragma unroll
      for (int r = 0; r < 8; ++r) acc[i][j][r] = 0.0f;

  for (int k0 = 0; k0 < CEMB; k0 += 32) {
    bf16x16 a0 = a_frag_bf16(yrow0, k0, lane);
    bf16x16 a1 = a_frag_bf16(yrow1, k0, lane);
#pragma unroll
    for (int j = 0; j < 4; ++j) {
      const __bf16* wcol = Wp + (size_t)(n0 + j * 16 + (lane & 15)) * CEMB;
      bf16x16 b = b_frag_bf16(wcol, k0, lane);
      acc[0][j] = wmma_bf16(a0, b, acc[0][j]);
      acc[1][j] = wmma_bf16(a1, b, acc[1][j]);
    }
  }

  int col  = lane & 15;
  int roff = (lane & 16) ? 8 : 0;
#pragma unroll
  for (int j = 0; j < 4; ++j) {
    int oc = n0 + j * 16 + col;
    float bias = Bp[oc];
#pragma unroll
    for (int i = 0; i < 2; ++i) {
#pragma unroll
      for (int r = 0; r < 8; ++r) {
        int bt = m0 + i * 16 + r + roff;
        Out[(size_t)bt * CEMB + oc] = acc[i][j][r] + bias;
      }
    }
  }
}

// ---------------------------------------------------------------------------
extern "C" void kernel_launch(void* const* d_in, const int* in_sizes, int n_in,
                              void* d_out, int out_size, void* d_ws, size_t ws_size,
                              hipStream_t stream) {
  const float* x      = (const float*)d_in[0];
  const float* w_attn = (const float*)d_in[1];
  const float* b_attn = (const float*)d_in[2];
  const float* w_proj = (const float*)d_in[3];
  const float* b_proj = (const float*)d_in[4];
  float* out = (float*)d_out;

  const size_t nX  = (size_t)BATCH * TSEQ * CEMB;        // 6,291,456
  const size_t nWa = (size_t)3 * CEMB * CEMB;            // 1,769,472
  const size_t nWp = (size_t)CEMB * CEMB;                //   589,824
  const size_t per = (size_t)BATCH * NHEAD * TSEQ * HD;  // 6,291,456

  __bf16* Xb  = (__bf16*)d_ws;
  __bf16* Wab = Xb + nX;
  __bf16* Wpb = Wab + nWa;
  __bf16* Qw  = Wpb + nWp;
  __bf16* Kw  = Qw + per;
  __bf16* Vt  = Kw + per;
  __bf16* Yw  = Vt + per;   // total ~67.6 MB bf16

  dim3 blk(128);
  cvt_kernel<<<dim3((unsigned)(nX  / 8 / 256)), 256, 0, stream>>>(x, Xb, (int)nX);
  cvt_kernel<<<dim3((unsigned)(nWa / 8 / 256)), 256, 0, stream>>>(w_attn, Wab, (int)nWa);
  cvt_kernel<<<dim3((unsigned)(nWp / 8 / 256)), 256, 0, stream>>>(w_proj, Wpb, (int)nWp);

  qkv_kernel<<<dim3(256, 9), blk, 0, stream>>>(Xb, Wab, b_attn, Qw, Kw, Vt);
  attn_kernel<<<dim3(BATCH * NHEAD * (TSEQ / 16) / 4), blk, 0, stream>>>(Qw, Kw, Vt, Yw);
  proj_kernel<<<dim3(256, 3), blk, 0, stream>>>(Yw, Wpb, b_proj, out);
}
